// Attention_12592844112346
// MI455X (gfx1250) — matrix operand order repaired and checked by reading, not compiled, not run on hardware
//
#include <hip/hip_runtime.h>
#include <hip/hip_bf16.h>

typedef __attribute__((ext_vector_type(16))) _Float16 v16h;
typedef __attribute__((ext_vector_type(8)))  _Float16 v8h;
typedef __attribute__((ext_vector_type(8)))  float    v8f;
typedef __attribute__((ext_vector_type(4)))  int      v4i;

#define WMMA_F16(a, b, c) \
  __builtin_amdgcn_wmma_f32_16x16x32_f16(false, (a), false, (b), (short)0, (c), false, false)

static __device__ __forceinline__ v16h cat8(v8h lo, v8h hi) {
  return __builtin_shufflevector(lo, hi, 0,1,2,3,4,5,6,7,8,9,10,11,12,13,14,15);
}

// Problem constants
#define BB 4
#define TT 2048
#define DD 2048
#define HH 16
#define DHD 128

// ---- CDNA5 async copy global->LDS (ASYNCcnt path), with sync fallback -----
#if __has_builtin(__builtin_amdgcn_global_load_async_to_lds_b128)
#define HAVE_ASYNC_LDS 1
#define ASYNC_CP_B128(gp, lp, off)                                        \
  __builtin_amdgcn_global_load_async_to_lds_b128(                         \
      (__attribute__((address_space(1))) v4i*)(gp),                       \
      (__attribute__((address_space(3))) v4i*)(lp), (off), 0)
#else
#define HAVE_ASYNC_LDS 0
#define ASYNC_CP_B128(gp, lp, off)                                        \
  do {                                                                    \
    *(v8h*)((_Float16*)(lp) + (off) / 2) =                                \
        *(const v8h*)((const _Float16*)(gp) + (off) / 2);                 \
  } while (0)
#endif

static __device__ __forceinline__ void wait_async_copies() {
#if HAVE_ASYNC_LDS
#if __has_builtin(__builtin_amdgcn_s_wait_asynccnt)
  __builtin_amdgcn_s_wait_asynccnt(0);
#else
  asm volatile("s_wait_asynccnt 0" ::: "memory");
#endif
#endif
}

// ---------------------------------------------------------------------------
// Elementwise fp32 -> f16 (contiguous), 8 elements per thread.
// ---------------------------------------------------------------------------
__global__ __launch_bounds__(256) void cvt_f16(const float* __restrict__ x,
                                               _Float16* __restrict__ y) {
  const size_t i = ((size_t)blockIdx.x * 256 + threadIdx.x) * 8;
  const float4 f0 = *(const float4*)(x + i);
  const float4 f1 = *(const float4*)(x + i + 4);
  v8h h;
  h[0] = (_Float16)f0.x; h[1] = (_Float16)f0.y;
  h[2] = (_Float16)f0.z; h[3] = (_Float16)f0.w;
  h[4] = (_Float16)f1.x; h[5] = (_Float16)f1.y;
  h[6] = (_Float16)f1.z; h[7] = (_Float16)f1.w;
  *(v8h*)(y + i) = h;
}

// ---------------------------------------------------------------------------
// Weight fp32 [K=2048, N=2048] -> f16 transposed [N, K] via 64x64 LDS tile.
// ---------------------------------------------------------------------------
__global__ __launch_bounds__(256) void cvt_transpose(const float* __restrict__ W,
                                                     _Float16* __restrict__ Wt) {
  __shared__ __align__(16) _Float16 tile[64][72];  // padded vs bank conflicts
  const int tid = threadIdx.x;
  const int r0 = tid >> 2;           // 0..63
  const int c0 = (tid & 3) * 16;     // 0,16,32,48
  const int rbase = blockIdx.y * 64;
  const int cbase = blockIdx.x * 64;

  const float* src = W + (size_t)(rbase + r0) * 2048 + cbase + c0;
  #pragma unroll
  for (int j = 0; j < 4; ++j) {
    const float4 f = *(const float4*)(src + j * 4);
    tile[r0][c0 + j * 4 + 0] = (_Float16)f.x;
    tile[r0][c0 + j * 4 + 1] = (_Float16)f.y;
    tile[r0][c0 + j * 4 + 2] = (_Float16)f.z;
    tile[r0][c0 + j * 4 + 3] = (_Float16)f.w;
  }
  __syncthreads();

  v8h w0, w1;
  #pragma unroll
  for (int j = 0; j < 8; ++j) w0[j] = tile[c0 + j][r0];
  #pragma unroll
  for (int j = 0; j < 8; ++j) w1[j] = tile[c0 + 8 + j][r0];
  _Float16* dst = Wt + (size_t)(cbase + r0) * 2048 + rbase + c0;
  *(v8h*)(dst)     = w0;
  *(v8h*)(dst + 8) = w1;
}

// ---------------------------------------------------------------------------
// Tiled GEMM (all-f16 operands): C[M=8192,N=2048] = Ah[M,2048] * Wt[N,2048]^T
// Staging is pure async b128 copies global->LDS (no VALU in the hot loop).
// MODE 0: C f16 written as Q/K layout [BH, T, 128]
// MODE 1: C f16 written transposed   [BH, 128, T]   (for V)
// MODE 2: C fp32 row-major           [M, 2048]      (final proj)
// Block = 256 threads = 8 waves (4m x 2n); block tile 128x64;
// wave tile 32x32 = 2x2 WMMA accumulators; BK=32 -> 4 WMMA per K-step.
// ---------------------------------------------------------------------------
template <int MODE>
__global__ __launch_bounds__(256) void gemm_wmma(const _Float16* __restrict__ Ah,
                                                 const _Float16* __restrict__ Wt,
                                                 void* __restrict__ Cp) {
  __shared__ __align__(16) _Float16 As[128 * 32];  // [m][k]  8 KB
  __shared__ __align__(16) _Float16 Bs[64 * 32];   // [n][k]  4 KB

  const int tid  = threadIdx.x;
  const int lane = tid & 31;
  const int wave = tid >> 5;
  const int wm   = wave >> 1;          // 0..3  (m)
  const int wn   = wave & 1;           // 0..1  (n)
  const int hi   = lane >> 4;
  const int ln   = lane & 15;
  const int mbase = blockIdx.y * 128;
  const int nbase = blockIdx.x * 64;

  v8f acc[2][2];
  #pragma unroll
  for (int im = 0; im < 2; ++im)
    #pragma unroll
    for (int in = 0; in < 2; ++in)
      #pragma unroll
      for (int i = 0; i < 8; ++i) acc[im][in][i] = 0.0f;

  const int ar = tid >> 1;             // 0..127  A stage row
  const int ak = (tid & 1) * 16;       // 0/16    A stage k-base (16 halves)
  const int br = tid >> 2;             // 0..63   B stage n-row
  const int bk8 = (tid & 3) * 8;       // 0..24   B stage k-base (8 halves)

  const _Float16* aSrc = Ah + (size_t)(mbase + ar) * DD + ak;
  const _Float16* bSrc = Wt + (size_t)(nbase + br) * DD + bk8;
  _Float16* aDst = As + ar * 32 + ak;
  _Float16* bDst = Bs + br * 32 + bk8;

  for (int kt = 0; kt < DD / 32; ++kt) {
    // ---- async stage A tile (128x32) and B tile (64n x 32k), both f16
    ASYNC_CP_B128(aSrc + kt * 32, aDst, 0);
    ASYNC_CP_B128(aSrc + kt * 32, aDst, 16);
    ASYNC_CP_B128(bSrc + kt * 32, bDst, 0);
    wait_async_copies();
    __syncthreads();

    // ---- fragments: A rows wm*32 + im*16 + ln; B cols wn*32 + in*16 + ln
    v16h af[2], bf[2];
    #pragma unroll
    for (int im = 0; im < 2; ++im) {
      const _Float16* ap = As + (wm * 32 + im * 16 + ln) * 32;
      af[im] = cat8(*(const v8h*)(ap + hi * 16), *(const v8h*)(ap + hi * 16 + 8));
    }
    #pragma unroll
    for (int in = 0; in < 2; ++in) {
      const _Float16* bp = Bs + (wn * 32 + in * 16 + ln) * 32 + hi * 16;
      bf[in] = cat8(*(const v8h*)(bp), *(const v8h*)(bp + 8));
    }
    #pragma unroll
    for (int im = 0; im < 2; ++im)
      #pragma unroll
      for (int in = 0; in < 2; ++in)
        acc[im][in] = WMMA_F16(af[im], bf[in], acc[im][in]);
    __syncthreads();
  }

  // ---- epilogue
  #pragma unroll
  for (int im = 0; im < 2; ++im) {
    #pragma unroll
    for (int in = 0; in < 2; ++in) {
      const int ncol = nbase + wn * 32 + in * 16 + ln;
      #pragma unroll
      for (int r = 0; r < 8; ++r) {
        const int row = mbase + wm * 32 + im * 16 + r + 8 * hi;
        if (MODE == 2) {
          ((float*)Cp)[(size_t)row * 2048 + ncol] = acc[im][in][r];
        } else {
          const int bb = row >> 11;        // / T
          const int t  = row & 2047;
          const int hh = ncol >> 7;        // / 128
          const int d  = ncol & 127;
          if (MODE == 0) {
            ((_Float16*)Cp)[(((size_t)(bb * HH + hh)) * TT + t) * DHD + d] =
                (_Float16)acc[im][in][r];
          } else {  // MODE 1: V transposed [BH,128,T]
            ((_Float16*)Cp)[(((size_t)(bb * HH + hh)) * DHD + d) * TT + t] =
                (_Float16)acc[im][in][r];
          }
        }
      }
    }
  }
}

// ---------------------------------------------------------------------------
// Flash attention: grid = B*H*(T/128) blocks of 256 threads (8 waves).
// Each wave owns a 16-query tile; online softmax over 32-key chunks.
// Q,K layout [BH,T,128] f16; Vt layout [BH,128,T] f16; out merged [B*T, H*128].
// ---------------------------------------------------------------------------
__global__ __launch_bounds__(256) void attn_wmma(const _Float16* __restrict__ Q,
                                                 const _Float16* __restrict__ K,
                                                 const _Float16* __restrict__ Vt,
                                                 _Float16* __restrict__ Om) {
  __shared__ __align__(16) _Float16 plds[8 * 16 * 32];  // per-wave 16x32 P tile

  const int lane = threadIdx.x & 31;
  const int wave = threadIdx.x >> 5;
  const int hi = lane >> 4;
  const int ln = lane & 15;
  const int bh = blockIdx.x >> 4;            // T/128 = 16 q-blocks per (b,h)
  const int qb = blockIdx.x & 15;
  const int qbase = qb * 128 + wave * 16;
  const int b = bh >> 4, h = bh & 15;
  _Float16* pw = plds + wave * 512;

  // ---- Q fragments (reused across all key chunks): 4 dk-chunks of 32
  v16h qf[4];
  {
    const _Float16* qrow = Q + ((size_t)bh * TT + qbase + ln) * DHD;
    #pragma unroll
    for (int kc = 0; kc < 4; ++kc) {
      v8h a0 = *(const v8h*)(qrow + kc * 32 + hi * 16);
      v8h a1 = *(const v8h*)(qrow + kc * 32 + hi * 16 + 8);
      qf[kc] = cat8(a0, a1);
    }
  }

  v8f o[8];
  #pragma unroll
  for (int d = 0; d < 8; ++d)
    #pragma unroll
    for (int i = 0; i < 8; ++i) o[d][i] = 0.0f;
  float mrow[8], lrow[8];
  #pragma unroll
  for (int r = 0; r < 8; ++r) { mrow[r] = -1e30f; lrow[r] = 0.0f; }

  const float scale = 0.08838834764831845f;  // 1/sqrt(128)
  const int nchunks = (qbase + 16 + 31) >> 5;

  for (int c = 0; c < nchunks; ++c) {
    const int kbase = c * 32;

    // ---- scores S[16 x 32] = Q * K^T  (two 16x16 tiles, 4 dk-chunks each)
    v8f s0, s1;
    #pragma unroll
    for (int i = 0; i < 8; ++i) { s0[i] = 0.0f; s1[i] = 0.0f; }
    {
      // B-fragment of K^T: lane col = key, 16 consecutive dk at hi*16
      const _Float16* k0 = K + ((size_t)bh * TT + kbase + ln) * DHD + hi * 16;
      const _Float16* k1 = k0 + 16 * DHD;
      #pragma unroll
      for (int kc = 0; kc < 4; ++kc) {
        v16h bf0 = cat8(*(const v8h*)(k0 + kc * 32),
                        *(const v8h*)(k0 + kc * 32 + 8));
        s0 = WMMA_F16(qf[kc], bf0, s0);
        v16h bf1 = cat8(*(const v8h*)(k1 + kc * 32),
                        *(const v8h*)(k1 + kc * 32 + 8));
        s1 = WMMA_F16(qf[kc], bf1, s1);
      }
    }

    // ---- scale + causal mask (C layout: row r+8*hi, col ln)
    #pragma unroll
    for (int r = 0; r < 8; ++r) {
      const int qi = qbase + r + 8 * hi;
      float v0 = s0[r] * scale;
      float v1 = s1[r] * scale;
      if (kbase + ln > qi)      v0 = -1e30f;
      if (kbase + 16 + ln > qi) v1 = -1e30f;
      s0[r] = v0; s1[r] = v1;
    }

    // ---- online softmax: per-row reduce across the 16-lane half
    float alpha[8];
    #pragma unroll
    for (int r = 0; r < 8; ++r) {
      float mx = fmaxf(s0[r], s1[r]);
      #pragma unroll
      for (int off = 1; off < 16; off <<= 1)
        mx = fmaxf(mx, __shfl_xor(mx, off, 32));
      const float mnew = fmaxf(mrow[r], mx);
      alpha[r] = __expf(mrow[r] - mnew);
      const float p0 = __expf(s0[r] - mnew);
      const float p1 = __expf(s1[r] - mnew);
      float rs = p0 + p1;
      #pragma unroll
      for (int off = 1; off < 16; off <<= 1)
        rs += __shfl_xor(rs, off, 32);
      lrow[r] = lrow[r] * alpha[r] + rs;
      mrow[r] = mnew;
      // C-layout -> LDS (row-major 16x32) for A-layout reload
      const int m = r + 8 * hi;
      pw[m * 32 + ln]      = (_Float16)p0;
      pw[m * 32 + 16 + ln] = (_Float16)p1;
    }
    asm volatile("s_wait_dscnt 0" ::: "memory");  // per-wave LDS transpose sync

    // ---- P as A-fragment (16x32 f16)
    v8h p0v = *(const v8h*)(pw + ln * 32 + hi * 16);
    v8h p1v = *(const v8h*)(pw + ln * 32 + hi * 16 + 8);
    v16h pfrag = cat8(p0v, p1v);

    // ---- O = O*alpha + P @ V  (8 dv-chunks of 16)
    const _Float16* vbase = Vt + (size_t)bh * DHD * TT + kbase + hi * 16;
    #pragma unroll
    for (int d = 0; d < 8; ++d) {
      #pragma unroll
      for (int r = 0; r < 8; ++r) o[d][r] *= alpha[r];
      const _Float16* vp = vbase + (size_t)(d * 16 + ln) * TT;
      v16h vfrag = cat8(*(const v8h*)(vp), *(const v8h*)(vp + 8));
      o[d] = WMMA_F16(pfrag, vfrag, o[d]);
    }
  }

  // ---- normalize and write merged [B*T, H*128] f16
  #pragma unroll
  for (int d = 0; d < 8; ++d) {
    #pragma unroll
    for (int r = 0; r < 8; ++r) {
      const int t = qbase + r + 8 * hi;
      const float val = o[d][r] / lrow[r];
      Om[((size_t)(b * TT + t)) * (HH * DHD) + h * DHD + d * 16 + ln] =
          (_Float16)val;
    }
  }
}

// ---------------------------------------------------------------------------
extern "C" void kernel_launch(void* const* d_in, const int* in_sizes, int n_in,
                              void* d_out, int out_size, void* d_ws,
                              size_t ws_size, hipStream_t stream) {
  const float* q  = (const float*)d_in[0];
  const float* k  = (const float*)d_in[1];
  const float* v  = (const float*)d_in[2];
  // d_in[3] = mask (causal; implemented analytically)
  const float* Wq = (const float*)d_in[4];
  const float* Wk = (const float*)d_in[5];
  const float* Wv = (const float*)d_in[6];
  const float* Wo = (const float*)d_in[7];
  float* out = (float*)d_out;

  char* ws = (char*)d_ws;
  const size_t U  = (size_t)BB * TT * DD * sizeof(_Float16);  // 32 MB
  const size_t WU = (size_t)DD * DD * sizeof(_Float16);       //  8 MB
  _Float16* Qh     = (_Float16*)(ws + 0 * U);
  _Float16* Kh     = (_Float16*)(ws + 1 * U);
  _Float16* Vt     = (_Float16*)(ws + 2 * U);
  _Float16* merged = (_Float16*)(ws + 3 * U);
  _Float16* qh     = (_Float16*)(ws + 4 * U);
  _Float16* kh     = (_Float16*)(ws + 5 * U);
  _Float16* vh     = (_Float16*)(ws + 6 * U);
  _Float16* Wqt    = (_Float16*)(ws + 7 * U + 0 * WU);
  _Float16* Wkt    = (_Float16*)(ws + 7 * U + 1 * WU);
  _Float16* Wvt    = (_Float16*)(ws + 7 * U + 2 * WU);
  _Float16* Wot    = (_Float16*)(ws + 7 * U + 3 * WU);

  dim3 blk(256);

  // ---- one-time precision/layout conversion passes
  const int nCvt = (BB * TT * DD) / (256 * 8);                // 8192 blocks
  cvt_f16<<<nCvt, blk, 0, stream>>>(q, qh);
  cvt_f16<<<nCvt, blk, 0, stream>>>(k, kh);
  cvt_f16<<<nCvt, blk, 0, stream>>>(v, vh);
  dim3 gT(2048 / 64, 2048 / 64);                              // (32, 32)
  cvt_transpose<<<gT, blk, 0, stream>>>(Wq, Wqt);
  cvt_transpose<<<gT, blk, 0, stream>>>(Wk, Wkt);
  cvt_transpose<<<gT, blk, 0, stream>>>(Wv, Wvt);
  cvt_transpose<<<gT, blk, 0, stream>>>(Wo, Wot);

  // ---- projections (WMMA GEMMs with async-LDS staging)
  dim3 gGemm(2048 / 64, (BB * TT) / 128);                     // (32, 64)
  gemm_wmma<0><<<gGemm, blk, 0, stream>>>(qh, Wqt, (void*)Qh);
  gemm_wmma<0><<<gGemm, blk, 0, stream>>>(kh, Wkt, (void*)Kh);
  gemm_wmma<1><<<gGemm, blk, 0, stream>>>(vh, Wvt, (void*)Vt);

  // ---- causal flash attention (WMMA)
  dim3 gAttn(BB * HH * (TT / 128));                           // 1024
  attn_wmma<<<gAttn, blk, 0, stream>>>(Qh, Kh, Vt, merged);

  // ---- output projection (WMMA GEMM, fp32 out)
  gemm_wmma<2><<<gGemm, blk, 0, stream>>>(merged, Wot, (void*)out);
}
